// NAFMultiheadNetwork_41008347742424
// MI455X (gfx1250) — matrix operand order repaired and checked by reading, not compiled, not run on hardware
//
#include <hip/hip_runtime.h>
#include <hip/hip_bf16.h>
#include <cstdint>

// Sizes from the reference
#define S_DIM 512
#define B_DIM 2048
#define T_DIM 64
#define F_DIM 128
#define H_DIM 16
#define NHEAD 3

typedef __attribute__((ext_vector_type(16))) _Float16 v16h;
typedef __attribute__((ext_vector_type(8)))  _Float16 v8h;
typedef __attribute__((ext_vector_type(2)))  _Float16 v2h;
typedef __attribute__((ext_vector_type(8)))  float    v8f;

union V16 { v16h v; v8h h[2]; };

// ---------------------------------------------------------------------------
// Encoder: out[(i*rows + r)*16 + h] = X[r,:] @ W[i,:,h] + bias[i,h]
// ---------------------------------------------------------------------------
__global__ void enc_kernel(const float* __restrict__ Xin,
                           const float* __restrict__ W,
                           const float* __restrict__ bias,
                           float* __restrict__ out, int rows) {
    int idx = blockIdx.x * blockDim.x + threadIdx.x;
    int total = NHEAD * rows * H_DIM;
    if (idx >= total) return;
    int h = idx & (H_DIM - 1);
    int tmp = idx >> 4;
    int r = tmp % rows;
    int i = tmp / rows;
    const float* x = Xin + (size_t)r * F_DIM;
    const float* w = W + ((size_t)i * F_DIM) * H_DIM + h;
    float acc = bias[i * H_DIM + h];
    #pragma unroll 8
    for (int f = 0; f < F_DIM; ++f) acc += x[f] * w[(size_t)f * H_DIM];
    out[idx] = acc;
}

// ---------------------------------------------------------------------------
// Stage-2 collapse: v[s,j,f] = sum_h treeW[j,f,h]*xe2[j,s,h];
//                   c[s,j]   = sum_h treeB[j,h]  *xe2[j,s,h]
// ---------------------------------------------------------------------------
__global__ void v_kernel(const float* __restrict__ tW,
                         const float* __restrict__ tB,
                         const float* __restrict__ xe2,
                         float* __restrict__ V, float* __restrict__ C) {
    int idx = blockIdx.x * blockDim.x + threadIdx.x;
    int total = S_DIM * NHEAD * F_DIM;
    if (idx >= total) return;
    int f = idx & (F_DIM - 1);
    int j = (idx >> 7) % NHEAD;
    int s = idx / (NHEAD * F_DIM);
    const float* xe = xe2 + ((size_t)j * S_DIM + s) * H_DIM;
    const float* w  = tW  + ((size_t)j * F_DIM + f) * H_DIM;
    float a = 0.f;
    #pragma unroll
    for (int h = 0; h < H_DIM; ++h) a += w[h] * xe[h];
    V[((size_t)s * NHEAD + j) * F_DIM + f] = a;
    if (f == 0) {
        float cc = 0.f;
        #pragma unroll
        for (int h = 0; h < H_DIM; ++h) cc += tB[j * H_DIM + h] * xe[h];
        C[s * NHEAD + j] = cc;
    }
}

// ---------------------------------------------------------------------------
// bgX (B,F) f32 -> bgXT (F,B) f16  (WMMA B-operand feed)
// ---------------------------------------------------------------------------
__global__ void tr_kernel(const float* __restrict__ bgX, _Float16* __restrict__ bgXT) {
    int idx = blockIdx.x * blockDim.x + threadIdx.x;
    if (idx >= B_DIM * F_DIM) return;
    int f = idx & (F_DIM - 1);
    int b = idx >> 7;
    bgXT[(size_t)f * B_DIM + b] = (_Float16)bgX[idx];
}

// ---------------------------------------------------------------------------
// Main fused kernel: one workgroup per (s, head).
//   1) dots[s,:] + global max + e16[b] = exp(dots-M) in LDS (f16)
//   2) K-loop (double-buffered A, one barrier per chunk):
//      A[t,b] = mask*e16 staged in LDS; C += A@bgX via WMMA f16;
//      Z[t] = A @ ones via one extra WMMA (nh==0 waves)
//   3) leaf[t,f] = C/Z in LDS; stage-2 tree softmax over T fused per j
// ---------------------------------------------------------------------------
__global__ __launch_bounds__(256) void naf_main_kernel(
        const uint8_t* __restrict__ nhot,
        const float*   __restrict__ Xenc,
        const float*   __restrict__ BGenc,
        const _Float16* __restrict__ bgXT,
        const float*   __restrict__ Vv,
        const float*   __restrict__ Cc,
        float*         __restrict__ ACC) {

    const int s    = blockIdx.x;
    const int head = blockIdx.y;
    const int tid  = threadIdx.x;
    const int lane = tid & 31;
    const int wave = tid >> 5;
    const int m    = wave & 3;   // M-tile (rows of T)
    const int nh   = wave >> 2;  // which half of F

    __shared__ _Float16 e16[B_DIM];             // 4 KB
    __shared__ _Float16 Alds[2][T_DIM * 40];    // 2x5 KB double buffer (80B rows)
    __shared__ float    leaf[T_DIM * F_DIM];    // 32 KB
    __shared__ float    zacc[T_DIM];
    __shared__ float    red[256];
    __shared__ float    xe_s[H_DIM];
    __shared__ float    vlds[F_DIM];
    __shared__ float    d2[T_DIM];
    __shared__ float    b2[T_DIM];

    // ---------- stage 0: dots + exp ----------
    if (tid < H_DIM) xe_s[tid] = Xenc[((size_t)head * S_DIM + s) * H_DIM + tid];
    __syncthreads();

    float dloc[8];
    {
        const float* bge = BGenc + (size_t)head * B_DIM * H_DIM;
        float lmax = -3.0e38f;
        #pragma unroll
        for (int k = 0; k < 8; ++k) {
            int b = tid * 8 + k;
            const float* br = bge + (size_t)b * H_DIM;
            float d = 0.f;
            #pragma unroll
            for (int h = 0; h < H_DIM; ++h) d += xe_s[h] * br[h];
            dloc[k] = d;
            lmax = fmaxf(lmax, d);
        }
        red[tid] = lmax;
        __syncthreads();
        for (int off = 128; off > 0; off >>= 1) {
            if (tid < off) red[tid] = fmaxf(red[tid], red[tid + off]);
            __syncthreads();
        }
        float M = red[0];
        #pragma unroll
        for (int k = 0; k < 8; ++k)
            e16[tid * 8 + k] = (_Float16)__expf(dloc[k] - M);
    }
    __syncthreads();  // e16 visible before first staging read

    // ---------- stage 1: C[t,f] = sum_b mask*e_b*bgX[b,f] via WMMA ----------
    const v8f vzero = {0.f, 0.f, 0.f, 0.f, 0.f, 0.f, 0.f, 0.f};
    v8f acc[4] = {vzero, vzero, vzero, vzero};
    v8f accz   = vzero;

    v16h onesv;
    #pragma unroll
    for (int i = 0; i < 16; ++i) onesv[i] = (_Float16)1.0f;

    const int trow   = m * 16 + (lane & 15);
    const int koff_a = (lane & 16) ? 8 : 0;    // f16 elems within A row
    const int koff_b = (lane & 16) ? 8 : 0;    // f16 elems within bgXT row
    // A-staging mapping: each thread handles 2 adjacent b columns x 4 t rows
    const int bp2 = tid & 15;   // b pair index: b = b0 + 2*bp2 + {0,1}
    const int tg  = tid >> 4;   // 0..15 -> t rows tg*4 .. tg*4+3

    for (int kc = 0; kc < B_DIM / 32; ++kc) {
        const int b0  = kc * 32;
        _Float16* Ab  = Alds[kc & 1];

        // B fragments: prefetch before the staging barrier (global only, no
        // LDS dependency) so L2 latency overlaps the mask staging.
        V16 bf[4];
        #pragma unroll
        for (int nt = 0; nt < 4; ++nt) {
            const int f = nh * 64 + nt * 16 + (lane & 15);
            const _Float16* bp = bgXT + (size_t)f * B_DIM + (size_t)(b0 + koff_b);
            bf[nt].h[0] = *(const v8h*)(bp);
            bf[nt].h[1] = *(const v8h*)(bp + 16);
        }

        // stage A tile (64 x 32 f16) into this chunk's buffer:
        // double-buffering makes this safe without a leading barrier —
        // waves can be at most one barrier apart, and the previous chunk
        // used the other buffer.
        {
            size_t offe = (((size_t)s * B_DIM) + (size_t)(b0 + 2 * bp2)) * T_DIM + (size_t)tg * 4;
            uint32_t me = *(const uint32_t*)(nhot + offe);          // b even, 4 t's
            uint32_t mo = *(const uint32_t*)(nhot + offe + T_DIM);  // b odd,  4 t's
            _Float16 eve = e16[b0 + 2 * bp2];
            _Float16 evo = e16[b0 + 2 * bp2 + 1];
            #pragma unroll
            for (int ii = 0; ii < 4; ++ii) {
                v2h pr;
                pr[0] = ((me >> (8 * ii)) & 0xffu) ? eve : (_Float16)0.0f;
                pr[1] = ((mo >> (8 * ii)) & 0xffu) ? evo : (_Float16)0.0f;
                *(v2h*)&Ab[(tg * 4 + ii) * 40 + 2 * bp2] = pr;
            }
        }
        __syncthreads();

        // A fragment: ISA 16-bit A 16x32 layout -> two contiguous 16B runs per lane
        V16 af;
        const _Float16* ap = &Ab[trow * 40 + koff_a];
        af.h[0] = *(const v8h*)(ap);
        af.h[1] = *(const v8h*)(ap + 16);

        // 4 N-tiles of 16 f-columns each; B was prefetched
        #pragma unroll
        for (int nt = 0; nt < 4; ++nt) {
            acc[nt] = __builtin_amdgcn_wmma_f32_16x16x32_f16(
                false, af.v, false, bf[nt].v, (short)0, acc[nt], false, false);
        }
        // Z[t] row sums via ones-matrix WMMA (wave-uniform branch: EXEC stays full)
        if (nh == 0) {
            accz = __builtin_amdgcn_wmma_f32_16x16x32_f16(
                false, af.v, false, onesv, (short)0, accz, false, false);
        }
    }
    __syncthreads();

    // scatter Z: C/D layout -> lane 0 holds N=0 col for M=r, lane 16 for M=r+8
    if (nh == 0 && (lane & 15) == 0) {
        int base = m * 16 + ((lane & 16) ? 8 : 0);
        #pragma unroll
        for (int r = 0; r < 8; ++r) zacc[base + r] = accz[r];
    }
    __syncthreads();

    // normalize into leaf LDS: lanes 0-15: M=r, lanes 16-31: M=r+8
    {
        const int tt0 = m * 16 + ((lane & 16) ? 8 : 0);
        float rz[8];
        #pragma unroll
        for (int r = 0; r < 8; ++r) rz[r] = __builtin_amdgcn_rcpf(zacc[tt0 + r]);
        #pragma unroll
        for (int nt = 0; nt < 4; ++nt) {
            #pragma unroll
            for (int r = 0; r < 8; ++r) {
                int ff = nh * 64 + nt * 16 + (lane & 15);
                leaf[(tt0 + r) * F_DIM + ff] = acc[nt][r] * rz[r];
            }
        }
    }
    __syncthreads();

    // ---------- stage 2: tree heads (softmax over T, fused) ----------
    float outreg = 0.f;
    for (int j = 0; j < NHEAD; ++j) {
        if (tid < F_DIM) vlds[tid] = Vv[((size_t)s * NHEAD + j) * F_DIM + tid];
        __syncthreads();
        if (tid < T_DIM) {
            float d = Cc[s * NHEAD + j];
            #pragma unroll 8
            for (int f = 0; f < F_DIM; ++f) d += leaf[tid * F_DIM + f] * vlds[f];
            d2[tid] = d;
        }
        __syncthreads();
        if (tid < T_DIM) {
            float mx = d2[0];
            #pragma unroll 8
            for (int t = 1; t < T_DIM; ++t) mx = fmaxf(mx, d2[t]);
            b2[tid] = __expf(d2[tid] - mx);
        }
        __syncthreads();
        if (tid < F_DIM) {
            float num = 0.f, den = 0.f;
            #pragma unroll 8
            for (int t = 0; t < T_DIM; ++t) {
                float bb = b2[t];
                den += bb;
                num += bb * leaf[t * F_DIM + tid];
            }
            outreg += num * __builtin_amdgcn_rcpf(den);
        }
        __syncthreads();
    }
    if (tid < F_DIM)
        ACC[((size_t)head * S_DIM + s) * F_DIM + tid] = outreg;
}

// ---------------------------------------------------------------------------
// Average over heads, scale 1/3, replicate into 4 output copies
// ---------------------------------------------------------------------------
__global__ void final_kernel(const float* __restrict__ ACC, float* __restrict__ out) {
    int idx = blockIdx.x * blockDim.x + threadIdx.x;
    const int N = S_DIM * F_DIM;
    if (idx >= N) return;
    float v = (ACC[idx] + ACC[N + idx] + ACC[2 * N + idx]) * (1.0f / 3.0f);
    out[idx] = v;
    out[N + idx] = v;
    out[2 * N + idx] = v;
    out[3 * N + idx] = v;
}

// ---------------------------------------------------------------------------
extern "C" void kernel_launch(void* const* d_in, const int* in_sizes, int n_in,
                              void* d_out, int out_size, void* d_ws, size_t ws_size,
                              hipStream_t stream) {
    const float*   X     = (const float*)d_in[0];
    const float*   bgX   = (const float*)d_in[1];
    // d_in[2] background_y: unused — first_leaf_y never reaches the output
    const uint8_t* nhot  = (const uint8_t*)d_in[3];   // jax bool -> 1 byte
    const float*   leafW = (const float*)d_in[4];
    const float*   leafB = (const float*)d_in[5];
    const float*   treeW = (const float*)d_in[6];
    const float*   treeB = (const float*)d_in[7];
    float* out = (float*)d_out;

    // workspace layout (floats), then f16 bgXT; needs ~2.7 MB
    float* wsf   = (float*)d_ws;
    float* Xenc  = wsf;                 // 3*512*16   = 24576
    float* BGenc = Xenc  + 24576;       // 3*2048*16  = 98304
    float* XE2   = BGenc + 98304;       // 24576
    float* Vv    = XE2   + 24576;       // 512*3*128  = 196608
    float* Cc    = Vv    + 196608;      // 512*3      = 1536
    float* ACC   = Cc    + 1536;        // 3*512*128  = 196608
    _Float16* bgXT = (_Float16*)(ACC + 196608); // 128*2048 f16

    // encoders
    enc_kernel<<<(NHEAD * S_DIM * H_DIM + 255) / 256, 256, 0, stream>>>(X,   leafW, leafB, Xenc,  S_DIM);
    enc_kernel<<<(NHEAD * B_DIM * H_DIM + 255) / 256, 256, 0, stream>>>(bgX, leafW, leafB, BGenc, B_DIM);
    enc_kernel<<<(NHEAD * S_DIM * H_DIM + 255) / 256, 256, 0, stream>>>(X,   treeW, treeB, XE2,   S_DIM);
    // stage-2 collapsed vectors
    v_kernel<<<(S_DIM * NHEAD * F_DIM + 255) / 256, 256, 0, stream>>>(treeW, treeB, XE2, Vv, Cc);
    // f16 transpose of background_X for WMMA B operand
    tr_kernel<<<(B_DIM * F_DIM + 255) / 256, 256, 0, stream>>>(bgX, bgXT);

    // fused attention + tree stage
    dim3 grid(S_DIM, NHEAD);
    naf_main_kernel<<<grid, 256, 0, stream>>>(nhot, Xenc, BGenc, bgXT, Vv, Cc, ACC);

    // reduce heads, replicate 4 outputs
    final_kernel<<<(S_DIM * F_DIM + 255) / 256, 256, 0, stream>>>(ACC, out);
}
